// ConvLSTMDecoder_43903155700034
// MI455X (gfx1250) — matrix operand order repaired and checked by reading, not compiled, not run on hardware
//
#include <hip/hip_runtime.h>
#include <stdint.h>

#define B_   64
#define T_   48
#define CIN  64
#define HID  128
#define HW   225      // 15*15
#define PADP 289      // 17*17
#define NTOT (B_*HW)  // 14400
#define FH   512      // 4*HID
#define KDNS (HID*HW) // 28800 dense K
#define DKCH 1920     // dense K per block -> 15 blocks, 60 chunks of 32
#define DNBLK 15

typedef __attribute__((ext_vector_type(16))) __bf16 v16bf;
typedef __attribute__((ext_vector_type(8)))  float  v8f;

union Frag { v16bf v; uint4 q[2]; };

__device__ __forceinline__ unsigned short f2bf(float f) {
    unsigned int u = __float_as_uint(f);
    u += 0x7FFFu + ((u >> 16) & 1u);           // round-to-nearest-even
    return (unsigned short)(u >> 16);
}
__device__ __forceinline__ float sigmoidf(float x) { return 1.0f / (1.0f + __expf(-x)); }
__device__ __forceinline__ float ftanhf(float x)   { return 1.0f - 2.0f / (1.0f + __expf(2.0f * x)); }

// ---------------- utility kernels ----------------
__global__ void zero_u32(uint32_t* p, long long n) {
    long long i = (long long)blockIdx.x * blockDim.x + threadIdx.x;
    if (i < n) p[i] = 0u;
}

// conv weights: W[o][c][ky][kx] f32 -> bf16 with K' = (ky*3+kx)*C + c  (window-major)
__global__ void conv_w_reorder(const float* __restrict__ s, unsigned short* __restrict__ d, int C) {
    int idx = blockIdx.x * 256 + threadIdx.x;
    int total = FH * C * 9;
    if (idx >= total) return;
    int o   = idx / (C * 9);
    int rem = idx - o * (C * 9);
    int c   = rem / 9;
    int rr  = rem - c * 9;
    d[(size_t)o * (C * 9) + rr * C + c] = f2bf(s[idx]);
}

// dense weights: Wd[K][128] f32 -> bf16 transposed [128][K]
__global__ void dense_w_tr(const float* __restrict__ s, unsigned short* __restrict__ d) {
    int idx = blockIdx.x * 256 + threadIdx.x;
    if (idx >= KDNS * 128) return;
    int k = idx >> 7, n = idx & 127;
    d[(size_t)n * KDNS + k] = f2bf(s[idx]);
}

// write x[:, t] (f32) into pad0 interior, channels 0..63, as bf16
__global__ void pack_x(const float* __restrict__ x, unsigned short* __restrict__ pad0, int t) {
    int idx = blockIdx.x * 256 + threadIdx.x;
    if (idx >= B_ * CIN * HW) return;
    int b   = idx / (CIN * HW);
    int rem = idx - b * (CIN * HW);
    int c   = rem / HW;
    int pos = rem - c * HW;
    int y = pos / 15, xx = pos - y * 15;
    float v = x[(((size_t)b * T_ + t) * CIN + c) * HW + pos];
    pad0[((size_t)b * 192 + c) * PADP + (y + 1) * 17 + (xx + 1)] = f2bf(v);
}

// ---------------- conv as implicit GEMM with WMMA ----------------
// z[M=512, N=14400] = Wbf[M, K'] * B[K', N] + bias; K' = (ky*3+kx)*C + c (window-major)
__global__ __launch_bounds__(256) void conv_wmma(
    const unsigned short* __restrict__ Wbf, const unsigned short* __restrict__ pad,
    const float* __restrict__ bias, float* __restrict__ z, int C, int Ktot)
{
    __shared__ unsigned short ldsA[2][128 * 40];   // [m][k] per 32-K chunk, stride 40
    __shared__ unsigned short ldsB[2][128 * 40];   // [n][k]

    const int tid  = threadIdx.x;
    const int mBlock = (blockIdx.x & 3) * 128;
    const int nBlock = (blockIdx.x >> 2) * 128;
    const int w = tid >> 5, lane = tid & 31;
    const int wm = w & 3, wn = w >> 2;             // 4x2 wave grid; wave tile 32x64

    v8f acc[2][4];
    #pragma unroll
    for (int i = 0; i < 2; ++i)
        #pragma unroll
        for (int j = 0; j < 4; ++j)
            acc[i][j] = (v8f){0.f,0.f,0.f,0.f,0.f,0.f,0.f,0.f};

    // B-staging: each thread owns one n column; k = k0 + 2r, address linear in k
    const int n_local = tid & 127;
    const int k0 = tid >> 7;
    const int ng = nBlock + n_local;
    const bool nvalid = ng < NTOT;
    int bb = 0, pos = 0;
    if (nvalid) { bb = ng / HW; pos = ng - bb * HW; }
    const int yy = pos / 15, xx = pos - (pos / 15) * 15;
    const unsigned short* padB = pad + (size_t)bb * C * PADP;

    // A-staging: row-linear in K'
    const int rowA = tid >> 1, partA = tid & 1;
    const unsigned short* gArow = Wbf + (size_t)(mBlock + rowA) * Ktot + partA * 16;

    const int nChunks = Ktot >> 5;

    #define STAGE_A(buf, kb) do {                                          \
        const uint4* ga = (const uint4*)(gArow + (kb));                    \
        uint4 a0 = ga[0], a1 = ga[1];                                      \
        uint4* da = (uint4*)&ldsA[(buf)][rowA * 40 + partA * 16];          \
        da[0] = a0; da[1] = a1;                                            \
    } while (0)

    #define STAGE_B(buf, cb, soff) do {                                    \
        const unsigned short* p = padB + (size_t)((cb) + k0) * PADP + (soff); \
        _Pragma("unroll")                                                  \
        for (int r = 0; r < 16; ++r) {                                     \
            unsigned short v = nvalid ? p[(size_t)(2 * r) * PADP] : (unsigned short)0; \
            ldsB[(buf)][n_local * 40 + (k0 + 2 * r)] = v;                  \
        }                                                                  \
    } while (0)

    int cb = 0, ky = 0, kx = 0;                    // state of the *staged* chunk
    STAGE_A(0, 0);
    STAGE_B(0, 0, yy * 17 + xx);

    for (int ch = 0; ch < nChunks; ++ch) {
        __syncthreads();
        if (ch + 1 < nChunks) {
            cb += 32;
            if (cb >= C) { cb = 0; if (++kx == 3) { kx = 0; ++ky; } }
            if (ch + 2 < nChunks) __builtin_prefetch(gArow + (ch + 2) * 32, 0, 0);
            STAGE_A((ch + 1) & 1, (ch + 1) * 32);
            STAGE_B((ch + 1) & 1, cb, (yy + ky) * 17 + (xx + kx));
        }
        const int p = ch & 1;

        Frag af[2], bf[4];
        const int kgA = lane >> 4;                 // K-half group of this lane
        #pragma unroll
        for (int i = 0; i < 2; ++i) {
            int row = wm * 32 + i * 16 + (lane & 15);
            af[i].q[0] = *(const uint4*)&ldsA[p][row * 40 + kgA * 8];       // K = kg*8..+7
            af[i].q[1] = *(const uint4*)&ldsA[p][row * 40 + 16 + kgA * 8];  // K = 16+kg*8..+7
        }
        #pragma unroll
        for (int j = 0; j < 4; ++j) {
            int col = wn * 64 + j * 16 + (lane & 15);
            bf[j].q[0] = *(const uint4*)&ldsB[p][col * 40 + kgA * 8];       // K = kg*8..+7
            bf[j].q[1] = *(const uint4*)&ldsB[p][col * 40 + 16 + kgA * 8];  // K = 16+kg*8..+7
        }
        #pragma unroll
        for (int i = 0; i < 2; ++i)
            #pragma unroll
            for (int j = 0; j < 4; ++j)
                acc[i][j] = __builtin_amdgcn_wmma_f32_16x16x32_bf16(
                    false, af[i].v, false, bf[j].v, (short)0, acc[i][j], false, false);
    }
    #undef STAGE_A
    #undef STAGE_B

    // store z = acc + bias ; C/D layout: VGPR v -> M = v + (lane>=16)*8, N = lane&15
    #pragma unroll
    for (int j = 0; j < 4; ++j) {
        int gn = nBlock + wn * 64 + j * 16 + (lane & 15);
        if (gn >= NTOT) continue;
        int b = gn / HW, pp = gn - (gn / HW) * HW;
        float* zb = z + ((size_t)b * FH) * HW + pp;
        #pragma unroll
        for (int i = 0; i < 2; ++i) {
            int m0 = mBlock + wm * 32 + i * 16 + ((lane >> 4) << 3);
            #pragma unroll
            for (int v = 0; v < 8; ++v) {
                int m = m0 + v;
                zb[(size_t)m * HW] = acc[i][j][v] + bias[m];
            }
        }
    }
}

// ---------------- LSTM gating ----------------
__global__ __launch_bounds__(256) void gate_kernel(
    const float* __restrict__ z, float* __restrict__ c,
    unsigned short* __restrict__ dst1, int C1, int ch1,
    unsigned short* __restrict__ dst2, int C2, int ch2, int dst2_padded)
{
    int idx = blockIdx.x * 256 + threadIdx.x;
    if (idx >= B_ * HID * HW) return;
    int b   = idx / (HID * HW);
    int rem = idx - b * (HID * HW);
    int hid = rem / HW;
    int pos = rem - hid * HW;
    const float* zp = z + ((size_t)b * FH + hid) * HW + pos;
    float i_ = sigmoidf(zp[0]);
    float f_ = sigmoidf(zp[(size_t)HID * HW]);
    float o_ = sigmoidf(zp[(size_t)2 * HID * HW]);
    float g_ = ftanhf(zp[(size_t)3 * HID * HW]);
    float cn = f_ * c[idx] + i_ * g_;
    c[idx] = cn;
    unsigned short hb = f2bf(o_ * ftanhf(cn));
    int y = pos / 15, xq = pos - y * 15;
    dst1[((size_t)b * C1 + ch1 + hid) * PADP + (y + 1) * 17 + (xq + 1)] = hb;
    if (dst2_padded)
        dst2[((size_t)b * C2 + ch2 + hid) * PADP + (y + 1) * 17 + (xq + 1)] = hb;
    else
        dst2[(size_t)b * (HID * HW) + hid * HW + pos] = hb;
}

// ---------------- dense head, K-split: partial[blk][64][128] = feat * Wdt over K chunk ----------------
__global__ __launch_bounds__(256) void dense_wmma(
    const unsigned short* __restrict__ feat, const unsigned short* __restrict__ Wdt,
    float* __restrict__ partial)
{
    __shared__ unsigned short ldsA[2][64 * 40];
    __shared__ unsigned short ldsB[2][128 * 40];
    const int tid = threadIdx.x, w = tid >> 5, lane = tid & 31;
    const int wm = w & 3, wn = w >> 2;             // wave tile 16x64
    const int kBase = blockIdx.x * DKCH;

    v8f acc[4];
    #pragma unroll
    for (int j = 0; j < 4; ++j) acc[j] = (v8f){0.f,0.f,0.f,0.f,0.f,0.f,0.f,0.f};

    const int rowA = tid >> 2, partA = tid & 3;    // 64 rows x 8 halves
    const int rowB = tid >> 1, partB = tid & 1;    // 128 rows x 16 halves
    const unsigned short* gA = feat + (size_t)rowA * KDNS + kBase + partA * 8;
    const unsigned short* gB = Wdt  + (size_t)rowB * KDNS + kBase + partB * 16;

    #define DSTAGE(buf, kb) do {                                           \
        *(uint4*)&ldsA[(buf)][rowA * 40 + partA * 8] = *(const uint4*)(gA + (kb)); \
        const uint4* gb = (const uint4*)(gB + (kb));                       \
        uint4 b0 = gb[0], b1 = gb[1];                                      \
        uint4* db = (uint4*)&ldsB[(buf)][rowB * 40 + partB * 16];          \
        db[0] = b0; db[1] = b1;                                            \
    } while (0)

    const int nChunks = DKCH >> 5;                 // 60
    DSTAGE(0, 0);
    for (int ch = 0; ch < nChunks; ++ch) {
        __syncthreads();
        if (ch + 1 < nChunks) DSTAGE((ch + 1) & 1, (ch + 1) * 32);
        const int p = ch & 1;

        Frag af, bf[4];
        int kg = lane >> 4;
        int row = wm * 16 + (lane & 15);
        af.q[0] = *(const uint4*)&ldsA[p][row * 40 + kg * 8];
        af.q[1] = *(const uint4*)&ldsA[p][row * 40 + 16 + kg * 8];
        #pragma unroll
        for (int j = 0; j < 4; ++j) {
            int col = wn * 64 + j * 16 + (lane & 15);
            bf[j].q[0] = *(const uint4*)&ldsB[p][col * 40 + kg * 8];
            bf[j].q[1] = *(const uint4*)&ldsB[p][col * 40 + 16 + kg * 8];
            acc[j] = __builtin_amdgcn_wmma_f32_16x16x32_bf16(
                false, af.v, false, bf[j].v, (short)0, acc[j], false, false);
        }
    }
    #undef DSTAGE

    float* po = partial + (size_t)blockIdx.x * (64 * 128);
    #pragma unroll
    for (int j = 0; j < 4; ++j) {
        int gn = wn * 64 + j * 16 + (lane & 15);
        int m0 = wm * 16 + ((lane >> 4) << 3);
        #pragma unroll
        for (int v = 0; v < 8; ++v)
            po[(size_t)(m0 + v) * 128 + gn] = acc[j][v];
    }
}

__global__ void dense_reduce(const float* __restrict__ partial, const float* __restrict__ bd,
                             float* __restrict__ out) {
    int idx = blockIdx.x * 256 + threadIdx.x;
    if (idx >= 64 * 128) return;
    float s = bd[idx & 127];
    #pragma unroll
    for (int i = 0; i < DNBLK; ++i) s += partial[(size_t)i * (64 * 128) + idx];
    out[idx] = s;
}

// ---------------- launcher ----------------
extern "C" void kernel_launch(void* const* d_in, const int* in_sizes, int n_in,
                              void* d_out, int out_size, void* d_ws, size_t ws_size,
                              hipStream_t stream) {
    const float* x  = (const float*)d_in[0];
    const float* W0 = (const float*)d_in[1];
    const float* b0 = (const float*)d_in[2];
    const float* W1 = (const float*)d_in[3];
    const float* b1 = (const float*)d_in[4];
    const float* Wd = (const float*)d_in[5];
    const float* bd = (const float*)d_in[6];
    (void)in_sizes; (void)n_in; (void)out_size; (void)ws_size;

    char* ws = (char*)d_ws;
    size_t off = 0;
    auto alloc = [&](size_t bytes) -> void* {
        off = (off + 255) & ~(size_t)255;
        void* p = ws + off;
        off += bytes;
        return p;
    };

    // zero-initialized region first (pads + cell states), then persistent data
    unsigned short* pad0    = (unsigned short*)alloc((size_t)B_ * 192 * PADP * 2);
    unsigned short* pad1    = (unsigned short*)alloc((size_t)B_ * 256 * PADP * 2);
    float*          c0      = (float*)alloc((size_t)B_ * HID * HW * 4);
    float*          c1      = (float*)alloc((size_t)B_ * HID * HW * 4);
    unsigned short* Wb0     = (unsigned short*)alloc((size_t)FH * 192 * 9 * 2);
    unsigned short* Wb1     = (unsigned short*)alloc((size_t)FH * 256 * 9 * 2);
    unsigned short* Wdt     = (unsigned short*)alloc((size_t)128 * KDNS * 2);
    unsigned short* hlast   = (unsigned short*)alloc((size_t)B_ * HID * HW * 2);
    float*          z       = (float*)alloc((size_t)B_ * FH * HW * 4);
    float*          partial = (float*)alloc((size_t)DNBLK * 64 * 128 * 4);

    // zero pads + cell states (one contiguous span)
    long long zn = ((char*)Wb0 - (char*)pad0) / 4;
    zero_u32<<<(unsigned)((zn + 255) / 256), 256, 0, stream>>>((uint32_t*)pad0, zn);

    // convert + reorder weights to bf16
    const int nW0 = FH * 192 * 9, nW1 = FH * 256 * 9, nWd = KDNS * 128;
    conv_w_reorder<<<(nW0 + 255) / 256, 256, 0, stream>>>(W0, Wb0, 192);
    conv_w_reorder<<<(nW1 + 255) / 256, 256, 0, stream>>>(W1, Wb1, 256);
    dense_w_tr<<<(nWd + 255) / 256, 256, 0, stream>>>(Wd, Wdt);

    const int convBlocks = 4 * ((NTOT + 127) / 128);   // 4 M-tiles x 113 N-tiles
    const int gateBlocks = (B_ * HID * HW) / 256;      // 7200
    const int packBlocks = (B_ * CIN * HW + 255) / 256;

    for (int t = 0; t < T_; ++t) {
        pack_x<<<packBlocks, 256, 0, stream>>>(x, pad0, t);
        // layer 0: input = pad0[ x_t (64ch) | h0 (128ch) ]
        conv_wmma<<<convBlocks, 256, 0, stream>>>(Wb0, pad0, b0, z, 192, 192 * 9);
        // h0 -> pad1 ch 0..127 (layer1 input now) and pad0 ch 64..191 (next step)
        gate_kernel<<<gateBlocks, 256, 0, stream>>>(z, c0, pad1, 256, 0, pad0, 192, 64, 1);
        // layer 1: input = pad1[ h0_t (128ch) | h1 (128ch) ]
        conv_wmma<<<convBlocks, 256, 0, stream>>>(Wb1, pad1, b1, z, 256, 256 * 9);
        // h1 -> pad1 ch 128..255 (next step) and hlast (dense input)
        gate_kernel<<<gateBlocks, 256, 0, stream>>>(z, c1, pad1, 256, 128, hlast, 0, 0, 0);
    }

    dense_wmma<<<DNBLK, 256, 0, stream>>>(hlast, Wdt, partial);
    dense_reduce<<<(64 * 128 + 255) / 256, 256, 0, stream>>>(partial, bd, (float*)d_out);
}
